// MoELayer_70549132804630
// MI455X (gfx1250) — matrix operand order repaired and checked by reading, not compiled, not run on hardware
//
#include <hip/hip_runtime.h>

typedef __attribute__((ext_vector_type(16))) __bf16 v16bf;
typedef __attribute__((ext_vector_type(8)))  float  v8f;

#define TOK_TILE 16
#define DDIM     1024
#define EDIM     256
#define XPAD     1028   // 1024 + 4 floats: row stride -> bank = (4*row + col) % 64
#define GPAD     260    // 256 + 4 floats

// exact (erf-based) GELU; erf via Abramowitz-Stegun 7.1.26 (|err| < 1.5e-7)
static __device__ __forceinline__ float gelu_erf(float z) {
    float u  = z * 0.70710678118654752f;     // z / sqrt(2)
    float a  = __builtin_fabsf(u);
    float tt = 1.0f / __builtin_fmaf(0.3275911f, a, 1.0f);   // v_rcp_f32
    float p  = __builtin_fmaf(tt, 1.061405429f, -1.453152027f);
    p        = __builtin_fmaf(tt, p, 1.421413741f);
    p        = __builtin_fmaf(tt, p, -0.284496736f);
    p        = __builtin_fmaf(tt, p, 0.254829592f);
    p       *= tt;
    float ex = __expf(-a * a);                               // v_exp_f32
    float er = __builtin_fmaf(-p, ex, 1.0f);                 // erf(|u|)
    er       = (u < 0.0f) ? -er : er;
    return 0.5f * z * (1.0f + er);
}

// Build a 16-bit A/B WMMA fragment (16 bf16 per lane) from 4 float4 chunks.
// Element order: [0..7] = first contiguous K-chunk, [8..15] = second K-chunk,
// matching the CDNA5 16-bit 16x32 A layout (and the mirrored 32x16 B layout).
static __device__ __forceinline__ v16bf frag16(float4 a0, float4 a1,
                                               float4 b0, float4 b1) {
    v16bf r;
    r[0]  = (__bf16)a0.x; r[1]  = (__bf16)a0.y; r[2]  = (__bf16)a0.z; r[3]  = (__bf16)a0.w;
    r[4]  = (__bf16)a1.x; r[5]  = (__bf16)a1.y; r[6]  = (__bf16)a1.z; r[7]  = (__bf16)a1.w;
    r[8]  = (__bf16)b0.x; r[9]  = (__bf16)b0.y; r[10] = (__bf16)b0.z; r[11] = (__bf16)b0.w;
    r[12] = (__bf16)b1.x; r[13] = (__bf16)b1.y; r[14] = (__bf16)b1.z; r[15] = (__bf16)b1.w;
    return r;
}

static __device__ __forceinline__ float4 ld4(const float* p) {
    return *(const float4*)p;
}

__global__ __launch_bounds__(256) void moe_fused_kernel(
    const float* __restrict__ x,   // [4096, 1024]
    const float* __restrict__ Wg,  // [256, 1024]
    const float* __restrict__ bg,  // [256]
    const float* __restrict__ W1,  // [256, 1024]
    const float* __restrict__ b1,  // [256]
    float* __restrict__ out)       // [4096, 256]
{
    __shared__ float xs[TOK_TILE][XPAD];     // x tile, f32
    __shared__ float gates[TOK_TILE][GPAD];  // logits -> softmax probs (in place)
    __shared__ float wbar[TOK_TILE][XPAD];   // gate-weighted W1 rows
    __shared__ float cvals[TOK_TILE];        // <gate, b1> per token

    const int tid  = threadIdx.x;
    const int wv   = tid >> 5;     // wave id 0..7
    const int lane = tid & 31;
    const int r    = lane & 15;    // M-row (A) / N-col (B) index within 16
    const int h    = lane >> 4;    // half selector
    const long t0  = (long)blockIdx.x * TOK_TILE;

    // ---------- Phase 0: async-stage x tile [16 x 1024] into LDS ----------
    // GLOBAL_LOAD_ASYNC_TO_LDS_B128: memory -> LDS directly (no VGPR data),
    // per-lane LDS destination address preserves the padded row stride.
    // Tracked by ASYNCcnt (CDNA5 split counter).
#pragma unroll
    for (int it = 0; it < (TOK_TILE * DDIM / 4) / 256; ++it) {
        int i = it * 256 + tid;
        int t = i >> 8;      // DDIM/4 == 256 float4 per row
        int c = i & 255;
        unsigned           lds_off = (unsigned)(unsigned long long)(const void*)&xs[t][c * 4];
        unsigned long long gaddr   = (unsigned long long)(const void*)(x + (t0 + t) * DDIM + c * 4);
        asm volatile("global_load_async_to_lds_b128 %0, %1, off"
                     :: "v"(lds_off), "v"(gaddr)
                     : "memory");
    }
    asm volatile("s_wait_asynccnt 0x0" ::: "memory");
    __syncthreads();

    // ---------- Phase 1: logits = x @ Wg^T + bg  (WMMA bf16, f32 acc) ----------
    {
        const int et0 = wv * 2, et1 = wv * 2 + 1;   // two 16-expert tiles per wave
        v8f acc0 = {};
        v8f acc1 = {};
        for (int kb = 0; kb < DDIM; kb += 32) {
            // A-frag: lane holds token row r, K-chunks [kb+8h, +8) and [kb+16+8h, +8)
            v16bf a = frag16(ld4(&xs[r][kb + h * 8]),      ld4(&xs[r][kb + h * 8 + 4]),
                             ld4(&xs[r][kb + 16 + h * 8]), ld4(&xs[r][kb + 16 + h * 8 + 4]));
            // B-frag: lane holds expert column (et*16 + r), K-chunks [kb+8h, +8) and [kb+16+8h, +8) from Wg row
            const float* wp0 = Wg + (long)(et0 * 16 + r) * DDIM + kb + h * 8;
            const float* wp1 = Wg + (long)(et1 * 16 + r) * DDIM + kb + h * 8;
            __builtin_prefetch(wp0 + 32, 0, 0);   // next kb -> global_prefetch_b8
            v16bf bA = frag16(ld4(wp0), ld4(wp0 + 4), ld4(wp0 + 16), ld4(wp0 + 20));
            v16bf bB = frag16(ld4(wp1), ld4(wp1 + 4), ld4(wp1 + 16), ld4(wp1 + 20));
            acc0 = __builtin_amdgcn_wmma_f32_16x16x32_bf16(false, a, false, bA,
                                                           (short)0, acc0, false, false);
            acc1 = __builtin_amdgcn_wmma_f32_16x16x32_bf16(false, a, false, bB,
                                                           (short)0, acc1, false, false);
        }
        const float bg0 = bg[et0 * 16 + r];
        const float bg1 = bg[et1 * 16 + r];
#pragma unroll
        for (int q = 0; q < 8; ++q) {   // C/D layout: VGPR q -> M = q + 8*h, N = r
            gates[q + 8 * h][et0 * 16 + r] = acc0[q] + bg0;
            gates[q + 8 * h][et1 * 16 + r] = acc1[q] + bg1;
        }
    }
    __syncthreads();

    // ---------- Phase 2: softmax over E per token + c[t] = <gate, b1> ----------
    {
        const int t = tid >> 4;         // token 0..15
        const int j = tid & 15;         // 16 threads per token, 16 experts each
        float v[16];
#pragma unroll
        for (int q = 0; q < 4; ++q) {
            float4 f = *(const float4*)&gates[t][j * 16 + q * 4];
            v[q * 4 + 0] = f.x; v[q * 4 + 1] = f.y; v[q * 4 + 2] = f.z; v[q * 4 + 3] = f.w;
        }
        float mx = v[0];
#pragma unroll
        for (int i = 1; i < 16; ++i) mx = fmaxf(mx, v[i]);
#pragma unroll
        for (int m = 1; m < 16; m <<= 1) mx = fmaxf(mx, __shfl_xor(mx, m, 16));
        float s = 0.0f;
#pragma unroll
        for (int i = 0; i < 16; ++i) { v[i] = __expf(v[i] - mx); s += v[i]; }
#pragma unroll
        for (int m = 1; m < 16; m <<= 1) s += __shfl_xor(s, m, 16);
        const float inv = 1.0f / s;
        float cpart = 0.0f;
#pragma unroll
        for (int i = 0; i < 16; ++i) {
            float ge = v[i] * inv;
            v[i] = ge;
            cpart = __builtin_fmaf(ge, b1[j * 16 + i], cpart);
        }
#pragma unroll
        for (int q = 0; q < 4; ++q) {
            float4 f; f.x = v[q*4+0]; f.y = v[q*4+1]; f.z = v[q*4+2]; f.w = v[q*4+3];
            *(float4*)&gates[t][j * 16 + q * 4] = f;
        }
#pragma unroll
        for (int m = 1; m < 16; m <<= 1) cpart += __shfl_xor(cpart, m, 16);
        if (j == 0) cvals[t] = cpart;
    }
    __syncthreads();

    // ---------- Phase 3: wbar = gate @ W1  (M=16 tokens, K=256, N=1024; WMMA) ----------
    {
        v16bf af[8];   // A-frags for all 8 K-steps (gate rows, reused across 8 N-tiles)
#pragma unroll
        for (int kk = 0; kk < 8; ++kk) {
            af[kk] = frag16(ld4(&gates[r][kk * 32 + h * 8]),
                            ld4(&gates[r][kk * 32 + h * 8 + 4]),
                            ld4(&gates[r][kk * 32 + 16 + h * 8]),
                            ld4(&gates[r][kk * 32 + 16 + h * 8 + 4]));
        }
        for (int nt = 0; nt < 8; ++nt) {        // wave covers d in [wv*128, wv*128+128)
            const int d0 = wv * 128 + nt * 16 + r;
            v8f acc = {};
#pragma unroll
            for (int kk = 0; kk < 8; ++kk) {
                const float* wp = W1 + (long)(kk * 32 + h * 8) * DDIM + d0; // column d0, L2-hot
                v16bf bfr;
#pragma unroll
                for (int jj = 0; jj < 16; ++jj) bfr[jj] = (__bf16)wp[(long)(jj + (jj >= 8 ? 8 : 0)) * DDIM];
                acc = __builtin_amdgcn_wmma_f32_16x16x32_bf16(false, af[kk], false, bfr,
                                                              (short)0, acc, false, false);
            }
#pragma unroll
            for (int q = 0; q < 8; ++q) wbar[q + 8 * h][d0] = acc[q];
        }
    }
    __syncthreads();

    // ---------- Phase 4: out[t,e] = c[t] + sum_d gelu(x[t,d]*g[t,e]) * wbar[t,d] ----------
    {
        const int e = tid;   // one expert column per thread; lanes coalesce on e
        for (int t = 0; t < TOK_TILE; ++t) {
            const float g  = gates[t][e];
            const float cv = cvals[t];
            const float4* xr = (const float4*)&xs[t][0];     // broadcast ds_load_b128
            const float4* wr = (const float4*)&wbar[t][0];   // broadcast ds_load_b128
            float acc = 0.0f;
            for (int q = 0; q < DDIM / 4; ++q) {
                float4 xv = xr[q];
                float4 wv4 = wr[q];
                acc = __builtin_fmaf(gelu_erf(xv.x * g), wv4.x, acc);
                acc = __builtin_fmaf(gelu_erf(xv.y * g), wv4.y, acc);
                acc = __builtin_fmaf(gelu_erf(xv.z * g), wv4.z, acc);
                acc = __builtin_fmaf(gelu_erf(xv.w * g), wv4.w, acc);
            }
            out[(t0 + t) * EDIM + e] = acc + cv;
        }
    }
}

extern "C" void kernel_launch(void* const* d_in, const int* in_sizes, int n_in,
                              void* d_out, int out_size, void* d_ws, size_t ws_size,
                              hipStream_t stream) {
    (void)in_sizes; (void)n_in; (void)d_ws; (void)ws_size; (void)out_size;
    const float* x  = (const float*)d_in[0];
    const float* Wg = (const float*)d_in[1];
    const float* bg = (const float*)d_in[2];
    const float* W1 = (const float*)d_in[3];
    const float* b1 = (const float*)d_in[4];
    float* out = (float*)d_out;

    dim3 grid(256);   // 4096 tokens / 16 per block
    dim3 block(256);  // 8 wave32
    hipLaunchKernelGGL(moe_fused_kernel, grid, block, 0, stream, x, Wg, bg, W1, b1, out);
}
